// GraphIpaFrameDenoiser_77214922047883
// MI455X (gfx1250) — matrix operand order repaired and checked by reading, not compiled, not run on hardware
//
#include <hip/hip_runtime.h>
#include <math.h>

#define NNODE 2048
#define NHEAD 12
#define ECHNK 8192
#define NE_SP 81920
#define NE_SQ 32768

typedef __bf16 bf16_t;
typedef __attribute__((ext_vector_type(16))) __bf16 v16bf;
typedef __attribute__((ext_vector_type(8)))  __bf16 v8bf;
typedef __attribute__((ext_vector_type(8)))  float  v8f;

#define DEV __device__ __forceinline__

DEV bf16_t f2bf(float f) {
  unsigned u = __builtin_bit_cast(unsigned, f);
  unsigned r = (u + 0x7FFFu + ((u >> 16) & 1u)) >> 16;
  return __builtin_bit_cast(bf16_t, (unsigned short)r);
}

DEV unsigned fenc(float f) {
  unsigned u = __builtin_bit_cast(unsigned, f);
  return (u & 0x80000000u) ? ~u : (u | 0x80000000u);
}
DEV float fdec(unsigned u) {
  unsigned b = (u & 0x80000000u) ? (u & 0x7FFFFFFFu) : ~u;
  return __builtin_bit_cast(float, b);
}

DEV void quat2rot(const float* q, float* R) {
  float w = q[0], x = q[1], y = q[2], z = q[3];
  R[0] = 1.f - 2.f * (y * y + z * z); R[1] = 2.f * (x * y - w * z); R[2] = 2.f * (x * z + w * y);
  R[3] = 2.f * (x * y + w * z); R[4] = 1.f - 2.f * (x * x + z * z); R[5] = 2.f * (y * z - w * x);
  R[6] = 2.f * (x * z - w * y); R[7] = 2.f * (y * z + w * x); R[8] = 1.f - 2.f * (x * x + y * y);
}

DEV v16bf ldA(const bf16_t* p) {
  union { v16bf v; v8bf h[2]; } u;
  u.h[0] = *(const v8bf*)(p);
  u.h[1] = *(const v8bf*)(p + 16);
  return u.v;
}

// ---------------- weight packing: W[K,N] f32 -> fragment-major bf16 -------------
// layout: [ntile][ktile][lane(32)][16 bf16]; lane holds col n = nt*16+(lane&15),
// k = kt*32 + (lane>>4)*8 + (t&7) + (t>>3)*16   (B-matrix per-lane K layout)
__global__ void pack_w_kernel(const float* __restrict__ W, bf16_t* __restrict__ Wp,
                              int K, int Nc, int KT, int NT) {
  int idx = blockIdx.x * 256 + threadIdx.x;
  int total = NT * KT * 512;
  if (idx >= total) return;
  int t = idx & 15;
  int lane = (idx >> 4) & 31;
  int tile = idx >> 9;
  int kt = tile % KT;
  int nt = tile / KT;
  int n = nt * 16 + (lane & 15);
  int k = kt * 32 + (lane >> 4) * 8 + (t & 7) + (t >> 3) * 16;
  float v = (k < K && n < Nc) ? W[(size_t)k * Nc + n] : 0.f;
  Wp[idx] = f2bf(v);
}

// ---------------- WMMA GEMM: C = act(A[M,K] @ W[K,N] + bias) --------------------
// block = 128 threads = 4 waves; each wave owns a 32(M) x 64(N) tile.
__global__ __launch_bounds__(128) void gemm_bf16_kernel(
    const bf16_t* __restrict__ A, const bf16_t* __restrict__ Wp,
    const float* __restrict__ bias, float* __restrict__ Cf,
    bf16_t* __restrict__ Cb, int M, int K, int Nn, int act, int wf, int wb) {
  const int wave = threadIdx.x >> 5;
  const int lane = threadIdx.x & 31;
  const int half = lane >> 4;
  const int l15 = lane & 15;
  const int m0 = (blockIdx.y * 4 + wave) << 5;   // 32 rows per wave
  if (m0 >= M) return;
  const int n0 = blockIdx.x << 6;
  const int KT = K >> 5;
  int nv = (Nn - n0 + 15) >> 4; if (nv > 4) nv = 4;

  v8f acc[2][4];
#pragma unroll
  for (int ti = 0; ti < 2; ++ti)
#pragma unroll
    for (int j = 0; j < 4; ++j)
#pragma unroll
      for (int r = 0; r < 8; ++r) acc[ti][j][r] = 0.f;

  const bf16_t* ar0 = A + (size_t)(m0 + l15) * K + half * 8;
  const bf16_t* ar1 = ar0 + (size_t)16 * K;
  const bf16_t* wb0 = Wp + ((size_t)(n0 >> 4) * KT * 32 + lane) * 16;
  const size_t nts = (size_t)KT * 512;

  if (nv == 4) {
    // branch-free hot loop: 8 WMMA per K-step
    for (int kt = 0; kt < KT; ++kt) {
      v16bf af0 = ldA(ar0 + kt * 32);
      v16bf af1 = ldA(ar1 + kt * 32);
      __builtin_prefetch(ar0 + (kt + 8) * 32, 0, 0);
      __builtin_prefetch(ar1 + (kt + 8) * 32, 0, 0);
      const bf16_t* wp = wb0 + (size_t)kt * 512;
#pragma unroll
      for (int j = 0; j < 4; ++j) {
        v16bf bj = *(const v16bf*)(wp + (size_t)j * nts);
        acc[0][j] = __builtin_amdgcn_wmma_f32_16x16x32_bf16(false, af0, false, bj, (short)0, acc[0][j], false, false);
        acc[1][j] = __builtin_amdgcn_wmma_f32_16x16x32_bf16(false, af1, false, bj, (short)0, acc[1][j], false, false);
      }
    }
  } else {
    for (int kt = 0; kt < KT; ++kt) {
      v16bf af0 = ldA(ar0 + kt * 32);
      v16bf af1 = ldA(ar1 + kt * 32);
      const bf16_t* wp = wb0 + (size_t)kt * 512;
#pragma unroll
      for (int j = 0; j < 4; ++j) {
        if (j < nv) {
          v16bf bj = *(const v16bf*)(wp + (size_t)j * nts);
          acc[0][j] = __builtin_amdgcn_wmma_f32_16x16x32_bf16(false, af0, false, bj, (short)0, acc[0][j], false, false);
          acc[1][j] = __builtin_amdgcn_wmma_f32_16x16x32_bf16(false, af1, false, bj, (short)0, acc[1][j], false, false);
        }
      }
    }
  }

#pragma unroll
  for (int ti = 0; ti < 2; ++ti) {
#pragma unroll
    for (int j = 0; j < 4; ++j) {
      if (j < nv) {
        int col = n0 + j * 16 + l15;
        float bv = bias ? bias[col] : 0.f;
#pragma unroll
        for (int r = 0; r < 8; ++r) {
          int row = m0 + ti * 16 + half * 8 + r;
          float v = acc[ti][j][r] + bv;
          if (act) v = v > 0.f ? v : 0.f;
          size_t o = (size_t)row * Nn + col;
          if (wf) Cf[o] = v;
          if (wb) Cb[o] = f2bf(v);
        }
      }
    }
  }
}

// ---------------- setup kernels --------------------------------------------------
__global__ __launch_bounds__(256) void center_kernel(const float* __restrict__ rig, float* __restrict__ center) {
  __shared__ float sx[256], sy[256], sz[256];
  int j = threadIdx.x;
  float ax = 0.f, ay = 0.f, az = 0.f;
  for (int i = j; i < NNODE; i += 256) { ax += rig[i * 7 + 4]; ay += rig[i * 7 + 5]; az += rig[i * 7 + 6]; }
  sx[j] = ax; sy[j] = ay; sz[j] = az; __syncthreads();
  for (int st = 128; st > 0; st >>= 1) {
    if (j < st) { sx[j] += sx[j + st]; sy[j] += sy[j + st]; sz[j] += sz[j + st]; }
    __syncthreads();
  }
  if (j == 0) { center[0] = sx[0] / NNODE; center[1] = sy[0] / NNODE; center[2] = sz[0] / NNODE; }
}

__global__ void rigid_init_kernel(const float* __restrict__ rig, const float* __restrict__ center,
                                  const unsigned char* __restrict__ xm, float* __restrict__ quat,
                                  float* __restrict__ X, float* __restrict__ trans, float* __restrict__ maskf) {
  int n = blockIdx.x * 256 + threadIdx.x;
  if (n >= NNODE) return;
  float q0 = rig[n * 7], q1 = rig[n * 7 + 1], q2 = rig[n * 7 + 2], q3 = rig[n * 7 + 3];
  float nr = sqrtf(q0 * q0 + q1 * q1 + q2 * q2 + q3 * q3) + 1e-8f;
  quat[n * 4] = q0 / nr; quat[n * 4 + 1] = q1 / nr; quat[n * 4 + 2] = q2 / nr; quat[n * 4 + 3] = q3 / nr;
  for (int i = 0; i < 3; ++i) {
    float x = rig[n * 7 + 4 + i] - center[i];
    X[n * 3 + i] = x;
    trans[n * 3 + i] = x * 0.1f;
  }
  maskf[n] = 1.f - (xm[n] ? 1.f : 0.f);
}

__global__ void time_l1_kernel(const float* __restrict__ tt, const float* __restrict__ W1,
                               const float* __restrict__ b1, float* __restrict__ te1) {
  int idx = blockIdx.x * 256 + threadIdx.x;
  if (idx >= NNODE * 128) return;
  int n = idx >> 7, j = idx & 127;
  float t = tt[n];
  float acc = b1[j];
  for (int k = 0; k < 64; ++k) {
    float w = expf(logf(1000.f) * (float)(k & 31) / 31.f);
    float ffk = (k < 32) ? sinf(t * w) : cosf(t * w);
    acc += ffk * W1[k * 128 + j];
  }
  te1[idx] = acc > 0.f ? acc : 0.f;
}

__global__ void time_l2_kernel(const float* __restrict__ te1, const float* __restrict__ W2,
                               const float* __restrict__ b2, float* __restrict__ te) {
  int idx = blockIdx.x * 256 + threadIdx.x;
  if (idx >= NNODE * 64) return;
  int n = idx >> 6, j = idx & 63;
  float acc = b2[j];
  const float* r = te1 + (size_t)n * 128;
  for (int k = 0; k < 128; ++k) acc += r[k] * W2[k * 64 + j];
  te[idx] = acc > 0.f ? acc : 0.f;
}

__global__ void emb_build_kernel(const int* __restrict__ residx, const float* __restrict__ te,
                                 const float* __restrict__ nm, bf16_t* __restrict__ A) {
  int idx = blockIdx.x * 256 + threadIdx.x;
  if (idx >= NNODE * 352) return;
  int n = idx / 352, j = idx - n * 352;
  float v;
  float rid = (float)residx[n];
  if (j < 128) {
    float f = expf(-(2.f * j) * logf(10000.f) / 256.f);
    v = sinf(rid * f);
  } else if (j < 256) {
    float f = expf(-(2.f * (j - 128)) * logf(10000.f) / 256.f);
    v = cosf(rid * f);
  } else if (j < 320) v = te[n * 64 + (j - 256)];
  else if (j == 320) v = nm[n];
  else v = 0.f;
  A[idx] = f2bf(v);
}

__global__ void mask_scale_kernel(float* __restrict__ s, const float* __restrict__ maskf,
                                  bf16_t* __restrict__ sb) {
  int idx = blockIdx.x * 256 + threadIdx.x;
  if (idx >= NNODE * 256) return;
  float v = s[idx] * maskf[idx >> 8];
  s[idx] = v;
  sb[idx] = f2bf(v);
}

__global__ void edge_feat_kernel(const int* __restrict__ ei, const float* __restrict__ X,
                                 const int* __restrict__ residx, float* __restrict__ z, int NE) {
  int e = blockIdx.x * 256 + threadIdx.x;
  if (e >= NE) return;
  int src = ei[e], dst = ei[NE + e];
  float dx = X[dst * 3] - X[src * 3];
  float dy = X[dst * 3 + 1] - X[src * 3 + 1];
  float dz = X[dst * 3 + 2] - X[src * 3 + 2];
  float d = sqrtf(dx * dx + dy * dy + dz * dz);
  float* ze = z + (size_t)e * 128;
  const float sig = 20.f / 64.f;
  for (int i = 0; i < 64; ++i) {
    float mu = 20.f * i / 63.f;
    float u = (d - mu) / sig;
    ze[i] = expf(-u * u);
  }
  float rd = (float)(residx[dst] - residx[src]);
  for (int i = 0; i < 32; ++i) {
    float f = expf(-(2.f * i) * logf(10000.f) / 64.f);
    ze[64 + i] = sinf(rd * f);
    ze[96 + i] = cosf(rd * f);
  }
}

// ---------------- IPA kernels ----------------------------------------------------
__global__ void compute_rot_kernel(const float* __restrict__ quat, float* __restrict__ rot) {
  int n = blockIdx.x * 256 + threadIdx.x;
  if (n >= NNODE) return;
  quat2rot(quat + n * 4, rot + n * 9);
}

__global__ void to_global_kernel(const float* __restrict__ pl, const float* __restrict__ rot,
                                 const float* __restrict__ tr, float* __restrict__ pg, int npts) {
  int idx = blockIdx.x * 256 + threadIdx.x;
  if (idx >= NNODE * npts) return;
  int n = idx / npts, i = idx - n * npts;
  const float* R = rot + n * 9;
  const float* p = pl + (size_t)n * npts * 3 + i * 3;
  float x = p[0], y = p[1], zc = p[2];
  float* o = pg + (size_t)n * npts * 3 + i * 3;
  o[0] = R[0] * x + R[1] * y + R[2] * zc + tr[n * 3 + 0];
  o[1] = R[3] * x + R[4] * y + R[5] * zc + tr[n * 3 + 1];
  o[2] = R[6] * x + R[7] * y + R[8] * zc + tr[n * 3 + 2];
}

__global__ void logits_kernel(const int* __restrict__ ei, const float* __restrict__ z,
                              const float* __restrict__ qv, const float* __restrict__ kvv,
                              const float* __restrict__ qpg, const float* __restrict__ kvpg,
                              const float* __restrict__ bw, const float* __restrict__ bbias,
                              const float* __restrict__ headw, const float* __restrict__ maskf,
                              float* __restrict__ logits, int NE) {
  int idx = blockIdx.x * 256 + threadIdx.x;
  if (idx >= NE * NHEAD) return;
  int e = idx / NHEAD, h = idx - e * NHEAD;
  int src = ei[e], dst = ei[NE + e];
  const float* q = qv + (size_t)dst * 192 + h * 16;
  const float* k = kvv + (size_t)src * 384 + h * 32;
  float sq = 0.f;
  for (int c = 0; c < 16; ++c) sq += q[c] * k[c];
  float bz = bbias[h];
  const float* ze = z + (size_t)e * 128;
  for (int c = 0; c < 128; ++c) bz += ze[c] * bw[c * NHEAD + h];
  float d2 = 0.f;
  const float* qp = qpg + (size_t)dst * 144 + h * 12;
  const float* kp = kvpg + (size_t)src * 432 + h * 36;
  for (int j = 0; j < 12; ++j) { float df = qp[j] - kp[j]; d2 += df * df; }
  float hw = headw[h];
  hw = hw > 0.f ? hw + log1pf(expf(-hw)) : log1pf(expf(hw));
  float lg = sq * 0.14433757f + 0.57735027f * bz - 0.5f * 0.13608276f * hw * d2
             + 1e9f * (maskf[src] - 1.f);
  logits[idx] = lg;
}

__global__ void init_mx_kernel(unsigned* __restrict__ mx) {
  int idx = blockIdx.x * 256 + threadIdx.x;
  if (idx >= NNODE * NHEAD) return;
  mx[idx] = 0x007FFFFFu;  // fenc(-inf)
}

__global__ void segmax_kernel(const int* __restrict__ ei, const float* __restrict__ logits,
                              unsigned* __restrict__ mx, int NE) {
  int idx = blockIdx.x * 256 + threadIdx.x;
  if (idx >= NE * NHEAD) return;
  int e = idx / NHEAD, h = idx - e * NHEAD;
  int dst = ei[NE + e];
  atomicMax(&mx[dst * NHEAD + h], fenc(logits[idx]));
}

__global__ void exp_kernel(const int* __restrict__ ei, float* __restrict__ logits,
                           const unsigned* __restrict__ mx, float* __restrict__ den, int NE) {
  int idx = blockIdx.x * 256 + threadIdx.x;
  if (idx >= NE * NHEAD) return;
  int e = idx / NHEAD, h = idx - e * NHEAD;
  int dst = ei[NE + e];
  float ex = expf(logits[idx] - fdec(mx[dst * NHEAD + h]));
  logits[idx] = ex;
  atomicAdd(&den[dst * NHEAD + h], ex);
}

__global__ void scatter_ov_kernel(const int* __restrict__ ei, const float* __restrict__ ex,
                                  const float* __restrict__ den, const float* __restrict__ kvv,
                                  const float* __restrict__ kvpg, float* __restrict__ o,
                                  float* __restrict__ optg, int NE) {
  int idx = blockIdx.x * 256 + threadIdx.x;
  if (idx >= NE * NHEAD) return;
  int e = idx / NHEAD, h = idx - e * NHEAD;
  int src = ei[e], dst = ei[NE + e];
  float a = ex[idx] / (den[dst * NHEAD + h] + 1e-9f);
  const float* v = kvv + (size_t)src * 384 + h * 32 + 16;
  for (int c = 0; c < 16; ++c) atomicAdd(&o[(size_t)dst * 192 + h * 16 + c], a * v[c]);
  const float* vp = kvpg + (size_t)src * 432 + h * 36 + 12;
  for (int j = 0; j < 24; ++j) atomicAdd(&optg[(size_t)dst * 288 + h * 24 + j], a * vp[j]);
}

__global__ void scatter_opair_kernel(const int* __restrict__ ei, const float* __restrict__ ex,
                                     const float* __restrict__ den, const float* __restrict__ z,
                                     float* __restrict__ opair, int NE) {
  int idx = blockIdx.x * 256 + threadIdx.x;
  if (idx >= NE * 128) return;
  int e = idx >> 7, c = idx & 127;
  int dst = ei[NE + e];
  float zc = z[(size_t)e * 128 + c];
  for (int h = 0; h < NHEAD; ++h) {
    float a = ex[e * NHEAD + h] / (den[dst * NHEAD + h] + 1e-9f);
    atomicAdd(&opair[(size_t)dst * 1536 + h * 128 + c], a * zc);
  }
}

__global__ void feats_kernel(const float* __restrict__ o, const float* __restrict__ optg,
                             const float* __restrict__ opair, const float* __restrict__ rot,
                             const float* __restrict__ trans, bf16_t* __restrict__ feats) {
  int idx = blockIdx.x * 256 + threadIdx.x;
  if (idx >= NNODE * NHEAD) return;
  int n = idx / NHEAD, h = idx - n * NHEAD;
  const float* R = rot + n * 9;
  float t0 = trans[n * 3], t1 = trans[n * 3 + 1], t2 = trans[n * 3 + 2];
  bf16_t* f = feats + (size_t)n * 2112;
  const float* on = o + (size_t)n * 192 + h * 16;
  for (int c = 0; c < 16; ++c) f[h * 16 + c] = f2bf(on[c]);
  const float* og = optg + (size_t)n * 288 + h * 24;
  for (int p = 0; p < 8; ++p) {
    float v0 = og[p * 3] - t0, v1 = og[p * 3 + 1] - t1, v2 = og[p * 3 + 2] - t2;
    float l0 = R[0] * v0 + R[3] * v1 + R[6] * v2;
    float l1 = R[1] * v0 + R[4] * v1 + R[7] * v2;
    float l2 = R[2] * v0 + R[5] * v1 + R[8] * v2;
    f[192 + (h * 8 + p) * 3 + 0] = f2bf(l0);
    f[192 + (h * 8 + p) * 3 + 1] = f2bf(l1);
    f[192 + (h * 8 + p) * 3 + 2] = f2bf(l2);
    f[480 + h * 8 + p] = f2bf(sqrtf(l0 * l0 + l1 * l1 + l2 * l2 + 1e-8f));
  }
  const float* op = opair + (size_t)n * 1536 + h * 128;
  for (int c = 0; c < 128; ++c) f[576 + h * 128 + c] = f2bf(op[c]);
}

__global__ __launch_bounds__(256) void add_ln_kernel(const float* __restrict__ s, const float* __restrict__ u,
                                                     const float* __restrict__ g, const float* __restrict__ b,
                                                     const float* __restrict__ maskf, int mask_in, int mask_out,
                                                     float* __restrict__ so, bf16_t* __restrict__ sb) {
  __shared__ float sm[256];
  int n = blockIdx.x, j = threadIdx.x;
  float mv = maskf[n];
  float x = s[(size_t)n * 256 + j] + u[(size_t)n * 256 + j] * (mask_in ? mv : 1.f);
  sm[j] = x; __syncthreads();
  for (int st = 128; st > 0; st >>= 1) { if (j < st) sm[j] += sm[j + st]; __syncthreads(); }
  float mean = sm[0] / 256.f; __syncthreads();
  float d = x - mean;
  sm[j] = d * d; __syncthreads();
  for (int st = 128; st > 0; st >>= 1) { if (j < st) sm[j] += sm[j + st]; __syncthreads(); }
  float var = sm[0] / 256.f;
  float y = d * rsqrtf(var + 1e-5f) * g[j] + b[j];
  if (mask_out) y *= mv;
  so[(size_t)n * 256 + j] = y;
  sb[(size_t)n * 256 + j] = f2bf(y);
}

// ---------------- backbone update ------------------------------------------------
__global__ void bb_upd_kernel(const float* __restrict__ s, const float* __restrict__ w,
                              const float* __restrict__ b, const float* __restrict__ nm,
                              float* __restrict__ upd) {
  int idx = blockIdx.x * 256 + threadIdx.x;
  if (idx >= NNODE * 6) return;
  int n = idx / 6, j = idx - n * 6;
  const float* sr = s + (size_t)n * 256;
  float dot = 0.f;
  for (int k = 0; k < 256; ++k) dot += sr[k] * w[k * 6 + j];
  float m = nm[n];
  upd[idx] = (m * dot + b[j]) * m;
}

__global__ void compose_kernel(float* __restrict__ quat, float* __restrict__ trans,
                               const float* __restrict__ upd) {
  int n = blockIdx.x * 256 + threadIdx.x;
  if (n >= NNODE) return;
  float q0 = quat[n * 4], q1 = quat[n * 4 + 1], q2 = quat[n * 4 + 2], q3 = quat[n * 4 + 3];
  const float* u = upd + n * 6;
  float b1 = u[0], b2 = u[1], b3 = u[2];
  float n0 = q0 - q1 * b1 - q2 * b2 - q3 * b3;
  float n1 = q0 * b1 + q1 + q2 * b3 - q3 * b2;
  float n2 = q0 * b2 - q1 * b3 + q2 + q3 * b1;
  float n3 = q0 * b3 + q1 * b2 - q2 * b1 + q3;
  float nr = sqrtf(n0 * n0 + n1 * n1 + n2 * n2 + n3 * n3) + 1e-8f;
  float qo[4] = { q0, q1, q2, q3 };
  float R[9]; quat2rot(qo, R);
  quat[n * 4] = n0 / nr; quat[n * 4 + 1] = n1 / nr; quat[n * 4 + 2] = n2 / nr; quat[n * 4 + 3] = n3 / nr;
  for (int i = 0; i < 3; ++i)
    trans[n * 3 + i] += R[i * 3] * u[3] + R[i * 3 + 1] * u[4] + R[i * 3 + 2] * u[5];
}

// ---------------- edge transition ------------------------------------------------
__global__ void hcat_kernel(const int* __restrict__ ei, const float* __restrict__ ne,
                            const float* __restrict__ z, float* __restrict__ hf,
                            bf16_t* __restrict__ hb, int e0, int NE) {
  int idx = blockIdx.x * 256 + threadIdx.x;
  if (idx >= ECHNK * 384) return;
  int r = idx / 384, j = idx - r * 384;
  int e = e0 + r;
  float v;
  if (j < 128) v = ne[(size_t)ei[e] * 128 + j];
  else if (j < 256) v = ne[(size_t)ei[NE + e] * 128 + (j - 128)];
  else v = z[(size_t)e * 128 + (j - 256)];
  hf[idx] = v;
  hb[idx] = f2bf(v);
}

__global__ void add_res_kernel(const float* __restrict__ t2f, const float* __restrict__ hf,
                               bf16_t* __restrict__ fb) {
  int idx = blockIdx.x * 256 + threadIdx.x;
  if (idx >= ECHNK * 384) return;
  fb[idx] = f2bf(t2f[idx] + hf[idx]);
}

__global__ __launch_bounds__(128) void ln_edge_kernel(const float* __restrict__ fin,
                                                      const float* __restrict__ g, const float* __restrict__ b,
                                                      float* __restrict__ z, int e0) {
  __shared__ float sm[128];
  int r = blockIdx.x, j = threadIdx.x;
  float x = fin[(size_t)r * 128 + j];
  sm[j] = x; __syncthreads();
  for (int st = 64; st > 0; st >>= 1) { if (j < st) sm[j] += sm[j + st]; __syncthreads(); }
  float mean = sm[0] / 128.f; __syncthreads();
  float d = x - mean;
  sm[j] = d * d; __syncthreads();
  for (int st = 64; st > 0; st >>= 1) { if (j < st) sm[j] += sm[j + st]; __syncthreads(); }
  float var = sm[0] / 128.f;
  z[(size_t)(e0 + r) * 128 + j] = d * rsqrtf(var + 1e-5f) * g[j] + b[j];
}

// ---------------- torsion & outputs ----------------------------------------------
__global__ void torsion_fin_kernel(const float* __restrict__ h, const float* __restrict__ w,
                                   const float* __restrict__ b, float* __restrict__ psi) {
  int n = blockIdx.x * 256 + threadIdx.x;
  if (n >= NNODE) return;
  const float* hr = h + (size_t)n * 256;
  float u0 = b[0], u1 = b[1];
  for (int k = 0; k < 256; ++k) { u0 += hr[k] * w[k * 2]; u1 += hr[k] * w[k * 2 + 1]; }
  float nr = sqrtf(u0 * u0 + u1 * u1) + 1e-8f;
  psi[n * 2] = u0 / nr;
  psi[n * 2 + 1] = u1 / nr;
}

__global__ void s_copy_kernel(const float* __restrict__ s, float* __restrict__ o) {
  int idx = blockIdx.x * 256 + threadIdx.x;
  if (idx < NNODE * 256) o[idx] = s[idx];
}

__global__ void finalize_kernel(const float* __restrict__ quat, const float* __restrict__ trans,
                                const float* __restrict__ center, const float* __restrict__ psi,
                                float* __restrict__ out_rig, float* __restrict__ out_bb) {
  int n = blockIdx.x * 256 + threadIdx.x;
  if (n >= NNODE) return;
  float q[4] = { quat[n * 4], quat[n * 4 + 1], quat[n * 4 + 2], quat[n * 4 + 3] };
  float R[9]; quat2rot(q, R);
  float tf[3];
  for (int i = 0; i < 3; ++i) tf[i] = trans[n * 3 + i] * 10.f + center[i];
  for (int i = 0; i < 4; ++i) out_rig[n * 7 + i] = q[i];
  for (int i = 0; i < 3; ++i) out_rig[n * 7 + 4 + i] = tf[i];
  float sp = psi[n * 2], cp = psi[n * 2 + 1];
  float loc[5][3] = { { -0.525f, 1.363f, 0.f }, { 0.f, 0.f, 0.f }, { 1.526f, 0.f, 0.f },
                      { 2.153f, cp * 1.062f, sp * 1.062f }, { -0.529f, -0.774f, -1.205f } };
  for (int a = 0; a < 5; ++a)
    for (int i = 0; i < 3; ++i)
      out_bb[(size_t)n * 15 + a * 3 + i] =
          R[i * 3] * loc[a][0] + R[i * 3 + 1] * loc[a][1] + R[i * 3 + 2] * loc[a][2] + tf[i];
}

// =================================================================================
static inline int gdiv(int a, int b) { return (a + b - 1) / b; }

extern "C" void kernel_launch(void* const* d_in, const int* in_sizes, int n_in,
                              void* d_out, int out_size, void* d_ws, size_t ws_size,
                              hipStream_t stream) {
  (void)in_sizes; (void)n_in; (void)out_size; (void)ws_size;
  const float* rig = (const float*)d_in[0];
  const float* t_in = (const float*)d_in[1];
  const float* nmsk = (const float*)d_in[2];
  const unsigned char* xmsk = (const unsigned char*)d_in[3];
  const int* residx = (const int*)d_in[4];
  const int* ei_sp = (const int*)d_in[5];
  const int* ei_sq = (const int*)d_in[6];
  auto leaf = [&](int i) -> const float* { return (const float*)d_in[i]; };
  auto LF = [&](int L, int o) { return 9 + L * 60 + o; };

  char* base = (char*)d_ws;
  size_t off = 0;
  auto alloc = [&](size_t bytes) -> void* {
    void* p = base + off;
    off = (off + bytes + 255) & ~((size_t)255);
    return p;
  };

  auto packW = [&](int li, int K, int Nc) -> bf16_t* {
    int Kp = (K + 31) & ~31, KT = Kp >> 5, NT = (Nc + 15) >> 4;
    int total = KT * NT * 512;
    bf16_t* p = (bf16_t*)alloc((size_t)total * sizeof(bf16_t));
    pack_w_kernel<<<gdiv(total, 256), 256, 0, stream>>>(leaf(li), p, K, Nc, KT, NT);
    return p;
  };
  auto gemm = [&](const bf16_t* A, const bf16_t* W, const float* bias, float* Cf, bf16_t* Cb,
                  int M, int K, int Nc, int act) {
    dim3 g(gdiv(Nc, 64), gdiv(M, 128));
    gemm_bf16_kernel<<<g, 128, 0, stream>>>(A, W, bias, Cf, Cb, M, K, Nc, act,
                                            Cf != nullptr, Cb != nullptr);
  };

  // ---- pack all GEMM weights ----
  bf16_t* embW = packW(8, 321, 256);
  struct IpaW { bf16_t *q, *kv, *qp, *kvp, *out; } ipw[4][2];
  struct EtW { bf16_t *init, *t1, *t2, *fin; } etw[4][2];
  bf16_t* ntw[4][3];
  for (int L = 0; L < 4; ++L) {
    for (int p = 0; p < 2; ++p) {
      int B = p == 0 ? 12 : 25;
      ipw[L][p].q = packW(LF(L, B + 10), 256, 192);
      ipw[L][p].kv = packW(LF(L, B + 4), 256, 384);
      ipw[L][p].qp = packW(LF(L, B + 12), 256, 144);
      ipw[L][p].kvp = packW(LF(L, B + 6), 256, 432);
      ipw[L][p].out = packW(LF(L, B + 8), 2112, 256);
      int BO = p == 0 ? 2 : 50;
      etw[L][p].fin = packW(LF(L, BO + 1), 384, 128);
      etw[L][p].init = packW(LF(L, BO + 3), 256, 128);
      etw[L][p].t1 = packW(LF(L, BO + 7), 384, 384);
      etw[L][p].t2 = packW(LF(L, BO + 9), 384, 384);
    }
    ntw[L][0] = packW(LF(L, 43), 256, 256);
    ntw[L][1] = packW(LF(L, 45), 256, 256);
    ntw[L][2] = packW(LF(L, 47), 256, 256);
  }
  bf16_t* torW1 = packW(256, 256, 256);
  bf16_t* torW2 = packW(258, 256, 256);

  // ---- activation / scratch buffers ----
  float* sbuf = (float*)alloc((size_t)NNODE * 256 * 4);
  bf16_t* sb = (bf16_t*)alloc((size_t)NNODE * 256 * 2);
  float* zsp = (float*)alloc((size_t)NE_SP * 128 * 4);
  float* zsq = (float*)alloc((size_t)NE_SQ * 128 * 4);
  float* quatb = (float*)alloc((size_t)NNODE * 4 * 4);
  float* transb = (float*)alloc((size_t)NNODE * 3 * 4);
  float* Xb = (float*)alloc((size_t)NNODE * 3 * 4);
  float* rotb = (float*)alloc((size_t)NNODE * 9 * 4);
  float* maskf = (float*)alloc((size_t)NNODE * 4);
  float* centerb = (float*)alloc(64);
  float* te1b = (float*)alloc((size_t)NNODE * 128 * 4);
  float* teb = (float*)alloc((size_t)NNODE * 64 * 4);
  bf16_t* embAb = (bf16_t*)alloc((size_t)NNODE * 352 * 2);
  float* qv = (float*)alloc((size_t)NNODE * 192 * 4);
  float* kvv = (float*)alloc((size_t)NNODE * 384 * 4);
  float* qpv = (float*)alloc((size_t)NNODE * 144 * 4);
  float* kvpv = (float*)alloc((size_t)NNODE * 432 * 4);
  float* qpg = (float*)alloc((size_t)NNODE * 144 * 4);
  float* kvpg = (float*)alloc((size_t)NNODE * 432 * 4);
  float* logitsb = (float*)alloc((size_t)NE_SP * NHEAD * 4);
  unsigned* mxb = (unsigned*)alloc((size_t)NNODE * NHEAD * 4);
  float* denb = (float*)alloc((size_t)NNODE * NHEAD * 4);
  float* ob = (float*)alloc((size_t)NNODE * 192 * 4);
  float* optgb = (float*)alloc((size_t)NNODE * 288 * 4);
  float* opairb = (float*)alloc((size_t)NNODE * 1536 * 4);
  bf16_t* featsb = (bf16_t*)alloc((size_t)NNODE * 2112 * 2);
  float* sub = (float*)alloc((size_t)NNODE * 256 * 4);
  float* neb = (float*)alloc((size_t)NNODE * 128 * 4);
  float* hcf = (float*)alloc((size_t)ECHNK * 384 * 4);
  bf16_t* hcb = (bf16_t*)alloc((size_t)ECHNK * 384 * 2);
  bf16_t* t1b = (bf16_t*)alloc((size_t)ECHNK * 384 * 2);
  float* t2f = (float*)alloc((size_t)ECHNK * 384 * 4);
  bf16_t* finb = (bf16_t*)alloc((size_t)ECHNK * 384 * 2);
  float* finf = (float*)alloc((size_t)ECHNK * 128 * 4);
  bf16_t* h1b = (bf16_t*)alloc((size_t)NNODE * 256 * 2);
  bf16_t* h2b = (bf16_t*)alloc((size_t)NNODE * 256 * 2);
  float* h3f = (float*)alloc((size_t)NNODE * 256 * 4);
  float* updb = (float*)alloc((size_t)NNODE * 6 * 4);
  float* th2f = (float*)alloc((size_t)NNODE * 256 * 4);

  // ---- setup ----
  center_kernel<<<1, 256, 0, stream>>>(rig, centerb);
  rigid_init_kernel<<<gdiv(NNODE, 256), 256, 0, stream>>>(rig, centerb, xmsk, quatb, Xb, transb, maskf);
  time_l1_kernel<<<gdiv(NNODE * 128, 256), 256, 0, stream>>>(t_in, leaf(250), leaf(249), te1b);
  time_l2_kernel<<<gdiv(NNODE * 64, 256), 256, 0, stream>>>(te1b, leaf(252), leaf(251), teb);
  emb_build_kernel<<<gdiv(NNODE * 352, 256), 256, 0, stream>>>(residx, teb, nmsk, embAb);
  gemm(embAb, embW, leaf(7), sbuf, nullptr, NNODE, 352, 256, 0);
  mask_scale_kernel<<<gdiv(NNODE * 256, 256), 256, 0, stream>>>(sbuf, maskf, sb);
  edge_feat_kernel<<<gdiv(NE_SP, 256), 256, 0, stream>>>(ei_sp, Xb, residx, zsp, NE_SP);
  edge_feat_kernel<<<gdiv(NE_SQ, 256), 256, 0, stream>>>(ei_sq, Xb, residx, zsq, NE_SQ);

  auto run_ipa = [&](int L, int p, const int* ei, float* z, int NE) {
    int B = p == 0 ? 12 : 25;
    compute_rot_kernel<<<gdiv(NNODE, 256), 256, 0, stream>>>(quatb, rotb);
    gemm(sb, ipw[L][p].q, leaf(LF(L, B + 9)), qv, nullptr, NNODE, 256, 192, 0);
    gemm(sb, ipw[L][p].kv, leaf(LF(L, B + 3)), kvv, nullptr, NNODE, 256, 384, 0);
    gemm(sb, ipw[L][p].qp, leaf(LF(L, B + 11)), qpv, nullptr, NNODE, 256, 144, 0);
    gemm(sb, ipw[L][p].kvp, leaf(LF(L, B + 5)), kvpv, nullptr, NNODE, 256, 432, 0);
    to_global_kernel<<<gdiv(NNODE * 48, 256), 256, 0, stream>>>(qpv, rotb, transb, qpg, 48);
    to_global_kernel<<<gdiv(NNODE * 144, 256), 256, 0, stream>>>(kvpv, rotb, transb, kvpg, 144);
    logits_kernel<<<gdiv(NE * NHEAD, 256), 256, 0, stream>>>(
        ei, z, qv, kvv, qpg, kvpg, leaf(LF(L, B + 1)), leaf(LF(L, B + 0)), leaf(LF(L, B + 2)),
        maskf, logitsb, NE);
    init_mx_kernel<<<gdiv(NNODE * NHEAD, 256), 256, 0, stream>>>(mxb);
    segmax_kernel<<<gdiv(NE * NHEAD, 256), 256, 0, stream>>>(ei, logitsb, mxb, NE);
    hipMemsetAsync(denb, 0, (size_t)NNODE * NHEAD * 4, stream);
    exp_kernel<<<gdiv(NE * NHEAD, 256), 256, 0, stream>>>(ei, logitsb, mxb, denb, NE);
    hipMemsetAsync(ob, 0, (size_t)NNODE * 192 * 4, stream);
    hipMemsetAsync(optgb, 0, (size_t)NNODE * 288 * 4, stream);
    hipMemsetAsync(opairb, 0, (size_t)NNODE * 1536 * 4, stream);
    scatter_ov_kernel<<<gdiv(NE * NHEAD, 256), 256, 0, stream>>>(ei, logitsb, denb, kvv, kvpg, ob, optgb, NE);
    scatter_opair_kernel<<<gdiv(NE * 128, 256), 256, 0, stream>>>(ei, logitsb, denb, z, opairb, NE);
    feats_kernel<<<gdiv(NNODE * NHEAD, 256), 256, 0, stream>>>(ob, optgb, opairb, rotb, transb, featsb);
    gemm(featsb, ipw[L][p].out, leaf(LF(L, B + 7)), sub, nullptr, NNODE, 2112, 256, 0);
    int gL = p == 0 ? 39 : 41, bL = p == 0 ? 38 : 40;
    add_ln_kernel<<<NNODE, 256, 0, stream>>>(sbuf, sub, leaf(LF(L, gL)), leaf(LF(L, bL)), maskf, 1, 0, sbuf, sb);
  };

  auto run_et = [&](int L, int p, const int* ei, float* z, int NE) {
    int BO = p == 0 ? 2 : 50;
    gemm(sb, etw[L][p].init, leaf(LF(L, BO + 2)), neb, nullptr, NNODE, 256, 128, 0);
    for (int e0 = 0; e0 < NE; e0 += ECHNK) {
      hcat_kernel<<<gdiv(ECHNK * 384, 256), 256, 0, stream>>>(ei, neb, z, hcf, hcb, e0, NE);
      gemm(hcb, etw[L][p].t1, leaf(LF(L, BO + 6)), nullptr, t1b, ECHNK, 384, 384, 1);
      gemm(t1b, etw[L][p].t2, leaf(LF(L, BO + 8)), t2f, nullptr, ECHNK, 384, 384, 1);
      add_res_kernel<<<gdiv(ECHNK * 384, 256), 256, 0, stream>>>(t2f, hcf, finb);
      gemm(finb, etw[L][p].fin, leaf(LF(L, BO + 0)), finf, nullptr, ECHNK, 384, 128, 0);
      ln_edge_kernel<<<ECHNK, 128, 0, stream>>>(finf, leaf(LF(L, BO + 5)), leaf(LF(L, BO + 4)), z, e0);
    }
  };

  // ---- layer loop ----
  for (int L = 0; L < 4; ++L) {
    run_ipa(L, 0, ei_sp, zsp, NE_SP);
    run_ipa(L, 1, ei_sq, zsq, NE_SQ);
    gemm(sb, ntw[L][0], leaf(LF(L, 42)), nullptr, h1b, NNODE, 256, 256, 1);
    gemm(h1b, ntw[L][1], leaf(LF(L, 44)), nullptr, h2b, NNODE, 256, 256, 1);
    gemm(h2b, ntw[L][2], leaf(LF(L, 46)), h3f, nullptr, NNODE, 256, 256, 0);
    add_ln_kernel<<<NNODE, 256, 0, stream>>>(sbuf, h3f, leaf(LF(L, 49)), leaf(LF(L, 48)), maskf, 0, 1, sbuf, sb);
    bb_upd_kernel<<<gdiv(NNODE * 6, 256), 256, 0, stream>>>(sbuf, leaf(LF(L, 1)), leaf(LF(L, 0)), nmsk, updb);
    compose_kernel<<<gdiv(NNODE, 256), 256, 0, stream>>>(quatb, transb, updb);
    run_et(L, 0, ei_sp, zsp, NE_SP);
    run_et(L, 1, ei_sq, zsq, NE_SQ);
  }

  // ---- torsion + outputs ----
  float* outp = (float*)d_out;
  gemm(sb, torW1, leaf(255), nullptr, h1b, NNODE, 256, 256, 1);
  gemm(h1b, torW2, leaf(257), th2f, nullptr, NNODE, 256, 256, 1);
  torsion_fin_kernel<<<gdiv(NNODE, 256), 256, 0, stream>>>(th2f, leaf(254), leaf(253), outp + 14336);
  s_copy_kernel<<<gdiv(NNODE * 256, 256), 256, 0, stream>>>(sbuf, outp + 18432);
  finalize_kernel<<<gdiv(NNODE, 256), 256, 0, stream>>>(quatb, transb, centerb, outp + 14336,
                                                        outp, outp + 542720);
}
